// EncoderBlock_39109972198038
// MI455X (gfx1250) — matrix operand order repaired and checked by reading, not compiled, not run on hardware
//
#include <hip/hip_runtime.h>
#include <hip/hip_bf16.h>
#include <math.h>

typedef __bf16 bf16_t;
typedef __attribute__((ext_vector_type(16))) __bf16 v16bf;
typedef __attribute__((ext_vector_type(8)))  __bf16 v8bf;
typedef __attribute__((ext_vector_type(8)))  float  v8f;

#define LNEPS 1e-5f
#define HEPS  1e-6f

// ---------------------------------------------------------------------------
// Batched WMMA GEMM:  C = act(alpha * A(bf16) x B(f32|bf16) + bias)
// A: M x K row-major (ldA).  B: K x N (ldB) or TRANSB -> N x K (ldB).
// MODE: 0 plain (z-strided), 1 split-heads, 2 merge-heads.
// Template-specialized so each variant carries no dead branches.
// ---------------------------------------------------------------------------
struct GemmArgs {
  const bf16_t* A; long sAb; int ldA;
  const float* Bf; const bf16_t* Bb; long sBb; int ldB;
  const float* bias;
  float* C;   long sCb;  int ldC;
  bf16_t* Cb; long sCbb; int ldCb;
  const float* addC;
  int M, N, K;
  float alpha;
  int headDim, H, T;
  int mode, act, transB, bbf16;   // host-side dispatch only
};

__device__ __forceinline__ float gelu_tanh(float x) {
  float u = 0.7978845608028654f * (x + 0.044715f * x * x * x);
  return 0.5f * x * (1.0f + tanhf(u));
}

template<int MODE, int ACT, int TRANSB, int B_BF16>
__global__ __launch_bounds__(128) void gemm_wmma_kernel(GemmArgs p) {
  __shared__ __align__(32) bf16_t sA[64][32];    // [m][k]
  __shared__ __align__(32) bf16_t sBt[64][32];   // [n][k]  (transposed tile)

  const int z = blockIdx.z;
  const bf16_t* A = p.A + (long)z * p.sAb;
  const float*  Bf = B_BF16 ? nullptr : p.Bf + (long)z * p.sBb;
  const bf16_t* Bb = B_BF16 ? p.Bb + (long)z * p.sBb : nullptr;

  const int bm = blockIdx.y * 64, bn = blockIdx.x * 64;
  const int tid = threadIdx.x;
  const int wave = tid >> 5, lane = tid & 31;
  const int wr = (wave >> 1) * 32, wc = (wave & 1) * 32;

  v8f acc[2][2];
  #pragma unroll
  for (int i = 0; i < 2; ++i)
    #pragma unroll
    for (int j = 0; j < 2; ++j)
      #pragma unroll
      for (int r = 0; r < 8; ++r) acc[i][j][r] = 0.0f;

  const bool fullM = (bm + 64 <= p.M);
  const bool fullN = (bn + 64 <= p.N);

  const int nk = (p.K + 31) / 32;
  for (int kc = 0; kc < nk; ++kc) {
    const int k0 = kc * 32;
    const bool fullK = (k0 + 32 <= p.K);

    // ---- stage A tile (64x32 = [m][k]) ----
    if (fullM && fullK) {
      for (int i = tid; i < 64 * 32; i += 128) {
        int r = i >> 5, c = i & 31;
        sA[r][c] = A[(long)(bm + r) * p.ldA + (k0 + c)];
      }
    } else {
      for (int i = tid; i < 64 * 32; i += 128) {
        int r = i >> 5, c = i & 31;
        int gm = bm + r, gk = k0 + c;
        bf16_t v = (bf16_t)0.0f;
        if (gm < p.M && gk < p.K) v = A[(long)gm * p.ldA + gk];
        sA[r][c] = v;
      }
    }
    // ---- stage B tile transposed ([n][k]) ----
    if (fullN && fullK) {
      for (int i = tid; i < 64 * 32; i += 128) {
        int n = i >> 5, k = i & 31;
        long idx = TRANSB ? ((long)(bn + n) * p.ldB + (k0 + k))
                          : ((long)(k0 + k) * p.ldB + (bn + n));
        float v = B_BF16 ? (float)Bb[idx] : Bf[idx];
        sBt[n][k] = (bf16_t)v;
      }
    } else {
      for (int i = tid; i < 64 * 32; i += 128) {
        int n = i >> 5, k = i & 31;
        int gn = bn + n, gk = k0 + k;
        float v = 0.0f;
        if (gn < p.N && gk < p.K) {
          long idx = TRANSB ? ((long)gn * p.ldB + gk) : ((long)gk * p.ldB + gn);
          v = B_BF16 ? (float)Bb[idx] : Bf[idx];
        }
        sBt[n][k] = (bf16_t)v;
      }
    }
    __syncthreads();

    // Fragment loads (wave32), vectorized from LDS, in contiguous k order shared by A and B:
    // A 16x32: lane m=lane&15; lanes<16 K 0..15, lanes>=16 K 16..31
    // B 32x16: lane n=lane&15; lanes<16 K 0..15, lanes>=16 K 16..31
    const int am = lane & 15, ak = (lane < 16) ? 0 : 16;
    const int bnl = lane & 15, bk = (lane < 16) ? 0 : 16;
    v16bf af[2], bfr[2];
    #pragma unroll
    for (int t = 0; t < 2; ++t) {
      const int rowOff = wr + t * 16 + am;
      af[t] = *(const v16bf*)&sA[rowOff][ak];
      const int colOff = wc + t * 16 + bnl;
      bfr[t] = *(const v16bf*)&sBt[colOff][bk];
    }
    #pragma unroll
    for (int ti = 0; ti < 2; ++ti)
      #pragma unroll
      for (int tj = 0; tj < 2; ++tj)
        acc[ti][tj] = __builtin_amdgcn_wmma_f32_16x16x32_bf16(
            false, af[ti], false, bfr[tj], (short)0, acc[ti][tj], false, false);
    __syncthreads();
  }

  // epilogue; C/D layout: VGPR r -> row r + 8*(lane>=16), col = lane&15
  const int cn = lane & 15, rbase = (lane >> 4) * 8;
  #pragma unroll
  for (int ti = 0; ti < 2; ++ti)
    #pragma unroll
    for (int tj = 0; tj < 2; ++tj)
      for (int r = 0; r < 8; ++r) {
        int ml = wr + ti * 16 + rbase + r;
        int nl = wc + tj * 16 + cn;
        int gm = bm + ml, gn = bn + nl;
        if (gm >= p.M || gn >= p.N) continue;
        float v = acc[ti][tj][r] * p.alpha;
        if (p.bias) v += p.bias[gn];
        if (ACT == 1) v = gelu_tanh(v);
        else if (ACT == 2) v = tanhf(v);
        if (MODE == 0) {
          if (p.addC) v += p.addC[(long)gm * p.ldC + gn];
          if (p.C)  p.C [(long)z * p.sCb  + (long)gm * p.ldC  + gn] = v;
          if (p.Cb) p.Cb[(long)z * p.sCbb + (long)gm * p.ldCb + gn] = (bf16_t)v;
        } else if (MODE == 1) {     // (b*T+t, he*hd+d) -> (b,he,t,d); z==0
          int b = gm / p.T, t = gm % p.T;
          int he = gn / p.headDim, hd = gn % p.headDim;
          long o = (((long)(b * p.H + he)) * p.T + t) * p.headDim + hd;
          if (p.C)  p.C[o]  = v;
          if (p.Cb) p.Cb[o] = (bf16_t)v;
        } else {                    // z=b*H+he, gm=t -> (b,t, he*hd+d)
          int b = z / p.H, he = z % p.H;
          long row = (long)b * p.T + gm;
          int col = he * p.headDim + gn;
          if (p.C)  p.C [row * p.ldC  + col] = v;
          if (p.Cb) p.Cb[row * p.ldCb + col] = (bf16_t)v;
        }
      }
}

// ---------------------------------------------------------------------------
// Elementwise / reduction kernels
// ---------------------------------------------------------------------------
__device__ __forceinline__ float acosh_f(float x) {
  return logf(x + sqrtf(x * x - 1.0f));
}

__global__ __launch_bounds__(256) void ln_kernel(const float* x, const float* g,
                                                 const float* b, bf16_t* out, int D) {
  __shared__ float sh[256];
  const long row = blockIdx.x;
  const float* xr = x + row * D;
  float s = 0.f, s2 = 0.f;
  for (int i = threadIdx.x; i < D; i += 256) { float v = xr[i]; s += v; s2 += v * v; }
  sh[threadIdx.x] = s; __syncthreads();
  for (int o = 128; o > 0; o >>= 1) { if (threadIdx.x < o) sh[threadIdx.x] += sh[threadIdx.x + o]; __syncthreads(); }
  float mean = sh[0] / (float)D; __syncthreads();
  sh[threadIdx.x] = s2; __syncthreads();
  for (int o = 128; o > 0; o >>= 1) { if (threadIdx.x < o) sh[threadIdx.x] += sh[threadIdx.x + o]; __syncthreads(); }
  float var = sh[0] / (float)D - mean * mean;
  float inv = rsqrtf(var + LNEPS);
  for (int i = threadIdx.x; i < D; i += 256)
    out[row * D + i] = (bf16_t)((xr[i] - mean) * inv * g[i] + b[i]);
}

__global__ __launch_bounds__(256) void softmax_kernel(const float* S, bf16_t* P, int L) {
  __shared__ float sh[256];
  const long row = blockIdx.x;
  const float* sr = S + row * (long)L;
  bf16_t* pr = P + row * (long)L;
  float m = -1e30f;
  for (int i = threadIdx.x; i < L; i += 256) m = fmaxf(m, sr[i]);
  sh[threadIdx.x] = m; __syncthreads();
  for (int o = 128; o > 0; o >>= 1) { if (threadIdx.x < o) sh[threadIdx.x] = fmaxf(sh[threadIdx.x], sh[threadIdx.x + o]); __syncthreads(); }
  m = sh[0]; __syncthreads();
  float s = 0.f;
  for (int i = threadIdx.x; i < L; i += 256) s += expf(sr[i] - m);
  sh[threadIdx.x] = s; __syncthreads();
  for (int o = 128; o > 0; o >>= 1) { if (threadIdx.x < o) sh[threadIdx.x] += sh[threadIdx.x + o]; __syncthreads(); }
  float inv = 1.0f / sh[0];
  for (int i = threadIdx.x; i < L; i += 256) pr[i] = (bf16_t)(expf(sr[i] - m) * inv);
}

// Lorentz softmax: a = softmax_k( -arccosh(clip(q0*k0 - <qs,ks>, 1+eps))^2 / tau )
__global__ __launch_bounds__(256) void lsoftmax_kernel(const float* inner, const float* qh,
                                                       const float* kh, float* aL, int T) {
  __shared__ float vals[512];
  __shared__ float sh[256];
  const long row = blockIdx.x;            // z*T + q
  const int z = (int)(row / T);
  const float* ir = inner + row * (long)T;
  const float q0 = qh[row * 33];
  for (int k = threadIdx.x; k < T; k += 256) {
    float k0 = kh[((long)z * T + k) * 33];
    float mink = ir[k] - q0 * k0;          // <qs,ks> - q0*k0
    float a = fmaxf(-mink, 1.0f + HEPS);
    float d = acosh_f(a);
    vals[k] = -d * d;                      // TAU = 1
  }
  __syncthreads();
  float m = -1e30f;
  for (int k = threadIdx.x; k < T; k += 256) m = fmaxf(m, vals[k]);
  sh[threadIdx.x] = m; __syncthreads();
  for (int o = 128; o > 0; o >>= 1) { if (threadIdx.x < o) sh[threadIdx.x] = fmaxf(sh[threadIdx.x], sh[threadIdx.x + o]); __syncthreads(); }
  m = sh[0]; __syncthreads();
  float s = 0.f;
  for (int k = threadIdx.x; k < T; k += 256) s += expf(vals[k] - m);
  sh[threadIdx.x] = s; __syncthreads();
  for (int o = 128; o > 0; o >>= 1) { if (threadIdx.x < o) sh[threadIdx.x] += sh[threadIdx.x + o]; __syncthreads(); }
  float inv = 1.0f / sh[0];
  for (int k = threadIdx.x; k < T; k += 256) aL[row * (long)T + k] = expf(vals[k] - m) * inv;
}

// lift: u(n,32) -> [sqrt(1+|u|^2), u] as f32 and bf16 (n,33)
__global__ __launch_bounds__(128) void lift_kernel(const float* u, float* ff, bf16_t* fb, int n) {
  int i = blockIdx.x * 128 + threadIdx.x;
  if (i >= n) return;
  const float* ur = u + (long)i * 32;
  float s = 1.0f;
  for (int j = 0; j < 32; ++j) s += ur[j] * ur[j];
  float t0 = sqrtf(s);
  long o = (long)i * 33;
  ff[o] = t0; fb[o] = (bf16_t)t0;
  for (int j = 0; j < 32; ++j) { ff[o + 1 + j] = ur[j]; fb[o + 1 + j] = (bf16_t)ur[j]; }
}

// per (z,q) row: alpha_k = clip(mu0*vh0_k - G, 1+eps); w1 = aL * acosh(alpha)/sqrt(alpha^2-1)
// w2 = sum_k w1*alpha
__global__ __launch_bounds__(256) void fweights_kernel(const float* G, const float* aL,
                                                       const float* mu, const float* vh,
                                                       bf16_t* w1, float* w2, int T) {
  __shared__ float sh[256];
  const long row = blockIdx.x;            // z*T + q
  const int z = (int)(row / T);
  const float mu0 = mu[row * 33];
  float acc = 0.f;
  for (int k = threadIdx.x; k < T; k += 256) {
    float vh0 = vh[((long)z * T + k) * 33];
    float alpha = fmaxf(mu0 * vh0 - G[row * (long)T + k], 1.0f + HEPS);
    float coef = acosh_f(alpha) * rsqrtf(alpha * alpha - 1.0f);
    float wv = aL[row * (long)T + k] * coef;
    w1[row * (long)T + k] = (bf16_t)wv;
    acc += wv * alpha;
  }
  sh[threadIdx.x] = acc; __syncthreads();
  for (int o = 128; o > 0; o >>= 1) { if (threadIdx.x < o) sh[threadIdx.x] += sh[threadIdx.x + o]; __syncthreads(); }
  if (threadIdx.x == 0) w2[row] = sh[0];
}

// exp map update: step = step_raw - w2*mu; v = step + <mu,step>_L mu; mu' = cosh(n)mu + sinh(n)/n v
__global__ __launch_bounds__(128) void expmap_kernel(const float* step_raw, const float* w2,
                                                     float* mu, bf16_t* mub, int n) {
  int i = blockIdx.x * 128 + threadIdx.x;
  if (i >= n) return;
  const float* sr = step_raw + (long)i * 33;
  float* m = mu + (long)i * 33;
  const float w = w2[i];
  float mm[33], st[33], v[33];
  for (int d = 0; d < 33; ++d) { mm[d] = m[d]; st[d] = sr[d] - w * mm[d]; }
  float ms = -mm[0] * st[0];
  for (int d = 1; d < 33; ++d) ms += mm[d] * st[d];
  for (int d = 0; d < 33; ++d) v[d] = st[d] + ms * mm[d];
  float vv = -v[0] * v[0];
  for (int d = 1; d < 33; ++d) vv += v[d] * v[d];
  float nn = sqrtf(fmaxf(vv, HEPS));
  float ch = coshf(nn), sh = sinhf(nn) / nn;
  for (int d = 0; d < 33; ++d) {
    float o = ch * mm[d] + sh * v[d];
    m[d] = o; mub[(long)i * 33 + d] = (bf16_t)o;
  }
}

// log map at origin, write merged (b, t, hh*33 + d) as bf16
__global__ __launch_bounds__(128) void logmap_kernel(const float* mu, bf16_t* y,
                                                     int T, int HH, int n) {
  int i = blockIdx.x * 128 + threadIdx.x;
  if (i >= n) return;                     // i over Z*T
  const int z = i / T, t = i % T;
  const int b = z / HH, hh = z % HH;
  const float* m = mu + (long)i * 33;
  float a = fmaxf(m[0], 1.0f + HEPS);
  float d = acosh_f(a);
  float coef = d * rsqrtf(a * a - 1.0f);
  bf16_t* yr = y + ((long)b * T + t) * (HH * 33) + hh * 33;
  yr[0] = (bf16_t)(coef * (m[0] - a));
  for (int dd = 1; dd < 33; ++dd) yr[dd] = (bf16_t)(coef * m[dd]);
}

// gate: alpha = sigmoid(dot(g_row, go_w)+go_b); x1 = x + alpha*hf + (1-alpha)*ef
__global__ __launch_bounds__(64) void gate_kernel(const float* g, const float* gow,
                                                  const float* gob, const float* ef,
                                                  const float* hf, const float* x,
                                                  float* x1, int GHn, int D) {
  __shared__ float sh[64];
  const long row = blockIdx.x;
  float v = (threadIdx.x < GHn) ? g[row * GHn + threadIdx.x] * gow[threadIdx.x] : 0.f;
  sh[threadIdx.x] = v; __syncthreads();
  for (int o = 32; o > 0; o >>= 1) { if (threadIdx.x < o) sh[threadIdx.x] += sh[threadIdx.x + o]; __syncthreads(); }
  float a = 1.0f / (1.0f + expf(-(sh[0] + gob[0])));
  for (int j = threadIdx.x; j < D; j += 64) {
    long o = row * D + j;
    x1[o] = x[o] + a * hf[o] + (1.0f - a) * ef[o];
  }
}

// ---------------------------------------------------------------------------
// Host orchestration
// ---------------------------------------------------------------------------
static GemmArgs ga0() { GemmArgs p{}; p.alpha = 1.0f; return p; }

static void launch_gemm(hipStream_t s, const GemmArgs& p, int batches) {
  dim3 g((p.N + 63) / 64, (p.M + 63) / 64, batches);
  dim3 b(128);
  // dispatch to the template instantiations actually used
  if (p.mode == 1)        hipLaunchKernelGGL((gemm_wmma_kernel<1,0,0,0>), g, b, 0, s, p);
  else if (p.mode == 2)   hipLaunchKernelGGL((gemm_wmma_kernel<2,0,0,1>), g, b, 0, s, p);
  else if (p.transB)      hipLaunchKernelGGL((gemm_wmma_kernel<0,0,1,1>), g, b, 0, s, p);
  else if (p.bbf16)       hipLaunchKernelGGL((gemm_wmma_kernel<0,0,0,1>), g, b, 0, s, p);
  else if (p.act == 1)    hipLaunchKernelGGL((gemm_wmma_kernel<0,1,0,0>), g, b, 0, s, p);
  else if (p.act == 2)    hipLaunchKernelGGL((gemm_wmma_kernel<0,2,0,0>), g, b, 0, s, p);
  else                    hipLaunchKernelGGL((gemm_wmma_kernel<0,0,0,0>), g, b, 0, s, p);
}

extern "C" void kernel_launch(void* const* d_in, const int* in_sizes, int n_in,
                              void* d_out, int out_size, void* d_ws, size_t ws_size,
                              hipStream_t stream) {
  const int Bc = 2, T = 512, D = 512, HE = 8, HD = 64, HH = 4, NL = 32;
  const int FFN = 2048, GH = 64;
  const int M = Bc * T;           // 1024
  const int ZE = Bc * HE;         // 16
  const int ZL = Bc * HH;         // 8
  const int LD33 = 33;

  const float* x     = (const float*)d_in[0];
  const float* ln1_g = (const float*)d_in[1];
  const float* ln1_b = (const float*)d_in[2];
  const float* wq = (const float*)d_in[3];  const float* bq = (const float*)d_in[4];
  const float* wk = (const float*)d_in[5];  const float* bk = (const float*)d_in[6];
  const float* wv = (const float*)d_in[7];  const float* bv = (const float*)d_in[8];
  const float* wo = (const float*)d_in[9];  const float* bo = (const float*)d_in[10];
  const float* lq_w = (const float*)d_in[11]; const float* lq_b = (const float*)d_in[12];
  const float* lk_w = (const float*)d_in[13]; const float* lk_b = (const float*)d_in[14];
  const float* lv_w = (const float*)d_in[15]; const float* lv_b = (const float*)d_in[16];
  const float* lo_w = (const float*)d_in[17]; const float* lo_b = (const float*)d_in[18];
  const float* ue_w = (const float*)d_in[19]; const float* ue_b = (const float*)d_in[20];
  const float* uh_w = (const float*)d_in[21]; const float* uh_b = (const float*)d_in[22];
  const float* gh_w = (const float*)d_in[23]; const float* gh_b = (const float*)d_in[24];
  const float* go_w = (const float*)d_in[25]; const float* go_b = (const float*)d_in[26];
  const float* ln2_g = (const float*)d_in[27]; const float* ln2_b = (const float*)d_in[28];
  const float* fi_w = (const float*)d_in[29]; const float* fi_b = (const float*)d_in[30];
  const float* fo_w = (const float*)d_in[31]; const float* fo_b = (const float*)d_in[32];
  float* out = (float*)d_out;

  char* base = (char*)d_ws; size_t off = 0;
  auto alloc = [&](size_t bytes) -> void* {
    void* p = base + off; off += (bytes + 255) & ~(size_t)255; return p;
  };

  bf16_t* h_bf   = (bf16_t*)alloc((size_t)M * D * 2);
  bf16_t* q_bf   = (bf16_t*)alloc((size_t)ZE * T * HD * 2);
  bf16_t* k_bf   = (bf16_t*)alloc((size_t)ZE * T * HD * 2);
  bf16_t* v_bf   = (bf16_t*)alloc((size_t)ZE * T * HD * 2);
  float*  S      = (float*) alloc((size_t)ZE * T * T * 4);   // also innerL | aL
  bf16_t* P      = (bf16_t*)alloc((size_t)ZE * T * T * 2);   // also w1
  bf16_t* attn_bf= (bf16_t*)alloc((size_t)M * D * 2);
  bf16_t* eout_bf= (bf16_t*)alloc((size_t)M * D * 2);
  float*  uq_f   = (float*) alloc((size_t)ZL * T * NL * 4);
  float*  uk_f   = (float*) alloc((size_t)ZL * T * NL * 4);
  float*  uv_f   = (float*) alloc((size_t)ZL * T * NL * 4);
  float*  qh_f   = (float*) alloc((size_t)ZL * T * LD33 * 4);
  float*  kh_f   = (float*) alloc((size_t)ZL * T * LD33 * 4);
  float*  vh_f   = (float*) alloc((size_t)ZL * T * LD33 * 4);
  bf16_t* qh_b   = (bf16_t*)alloc((size_t)ZL * T * LD33 * 2);
  bf16_t* kh_b   = (bf16_t*)alloc((size_t)ZL * T * LD33 * 2);
  bf16_t* vh_b   = (bf16_t*)alloc((size_t)ZL * T * LD33 * 2);
  float*  mu_f   = (float*) alloc((size_t)ZL * T * LD33 * 4);
  bf16_t* mu_b   = (bf16_t*)alloc((size_t)ZL * T * LD33 * 2);
  float*  w2_f   = (float*) alloc((size_t)ZL * T * 4);
  float*  step_f = (float*) alloc((size_t)ZL * T * LD33 * 4);
  bf16_t* y_bf   = (bf16_t*)alloc((size_t)M * (HH * LD33) * 2);
  bf16_t* hout_bf= (bf16_t*)alloc((size_t)M * D * 2);
  float*  ef_f   = (float*) alloc((size_t)M * D * 4);
  float*  hf_f   = (float*) alloc((size_t)M * D * 4);
  bf16_t* cat_bf = (bf16_t*)alloc((size_t)M * 2 * D * 2);
  float*  gate_f = (float*) alloc((size_t)M * GH * 4);
  float*  x1_f   = (float*) alloc((size_t)M * D * 4);
  bf16_t* h2_bf  = (bf16_t*)alloc((size_t)M * D * 2);
  bf16_t* ffn1_bf= (bf16_t*)alloc((size_t)M * FFN * 2);
  (void)ws_size; (void)n_in; (void)in_sizes; (void)out_size;

  float*  innerL = S;                         // alias (ZL*T*T floats)
  float*  aL     = S + (size_t)ZL * T * T;    // alias (second half of S region)
  bf16_t* w1_b   = P;                         // alias (ZL*T*T bf16)

  // ---- LN1 ----
  hipLaunchKernelGGL(ln_kernel, dim3(M), dim3(256), 0, stream, x, ln1_g, ln1_b, h_bf, D);

  // ---- Euclid QKV projections (split heads) ----
  for (int i = 0; i < 3; ++i) {
    GemmArgs p = ga0();
    p.A = h_bf; p.ldA = D;
    p.Bf = (i == 0 ? wq : i == 1 ? wk : wv); p.ldB = D;
    p.bias = (i == 0 ? bq : i == 1 ? bk : bv);
    p.Cb = (i == 0 ? q_bf : i == 1 ? k_bf : v_bf);
    p.M = M; p.N = D; p.K = D;
    p.mode = 1; p.headDim = HD; p.H = HE; p.T = T;
    launch_gemm(stream, p, 1);
  }

  // ---- Lorentz projections (split heads, f32 out) ----
  for (int i = 0; i < 3; ++i) {
    GemmArgs p = ga0();
    p.A = h_bf; p.ldA = D;
    p.Bf = (i == 0 ? lq_w : i == 1 ? lk_w : lv_w); p.ldB = HH * NL;
    p.bias = (i == 0 ? lq_b : i == 1 ? lk_b : lv_b);
    p.C = (i == 0 ? uq_f : i == 1 ? uk_f : uv_f);
    p.M = M; p.N = HH * NL; p.K = D;
    p.mode = 1; p.headDim = NL; p.H = HH; p.T = T;
    launch_gemm(stream, p, 1);
  }

  // ---- Euclid attention: scores, softmax, PV, Wo ----
  {
    GemmArgs p = ga0();                       // S = Q K^T / sqrt(HD)
    p.A = q_bf; p.sAb = (long)T * HD; p.ldA = HD;
    p.Bb = k_bf; p.sBb = (long)T * HD; p.ldB = HD; p.transB = 1; p.bbf16 = 1;
    p.C = S; p.sCb = (long)T * T; p.ldC = T;
    p.M = T; p.N = T; p.K = HD; p.alpha = 0.125f;
    launch_gemm(stream, p, ZE);
  }
  hipLaunchKernelGGL(softmax_kernel, dim3(ZE * T), dim3(256), 0, stream, S, P, T);
  {
    GemmArgs p = ga0();                       // O = P V, merged heads
    p.A = P; p.sAb = (long)T * T; p.ldA = T;
    p.Bb = v_bf; p.sBb = (long)T * HD; p.ldB = HD; p.bbf16 = 1;
    p.Cb = attn_bf; p.ldCb = D;
    p.M = T; p.N = HD; p.K = T;
    p.mode = 2; p.headDim = HD; p.H = HE; p.T = T;
    launch_gemm(stream, p, ZE);
  }
  {
    GemmArgs p = ga0();                       // e_out = O wo + bo
    p.A = attn_bf; p.ldA = D;
    p.Bf = wo; p.ldB = D; p.bias = bo;
    p.Cb = eout_bf; p.ldCb = D;
    p.M = M; p.N = D; p.K = D;
    launch_gemm(stream, p, 1);
  }

  // ---- Lorentz: lift, scores, softmax ----
  const int nZT = ZL * T;
  hipLaunchKernelGGL(lift_kernel, dim3((nZT + 127) / 128), dim3(128), 0, stream, uq_f, qh_f, qh_b, nZT);
  hipLaunchKernelGGL(lift_kernel, dim3((nZT + 127) / 128), dim3(128), 0, stream, uk_f, kh_f, kh_b, nZT);
  hipLaunchKernelGGL(lift_kernel, dim3((nZT + 127) / 128), dim3(128), 0, stream, uv_f, vh_f, vh_b, nZT);
  (void)hipMemcpyAsync(mu_f, qh_f, (size_t)nZT * LD33 * 4, hipMemcpyDeviceToDevice, stream);
  (void)hipMemcpyAsync(mu_b, qh_b, (size_t)nZT * LD33 * 2, hipMemcpyDeviceToDevice, stream);
  {
    GemmArgs p = ga0();                       // spatial inner <qs,ks>
    p.A = qh_b + 1; p.sAb = (long)T * LD33; p.ldA = LD33;
    p.Bb = kh_b + 1; p.sBb = (long)T * LD33; p.ldB = LD33; p.transB = 1; p.bbf16 = 1;
    p.C = innerL; p.sCb = (long)T * T; p.ldC = T;
    p.M = T; p.N = T; p.K = NL;
    launch_gemm(stream, p, ZL);
  }
  hipLaunchKernelGGL(lsoftmax_kernel, dim3(ZL * T), dim3(256), 0, stream, innerL, qh_f, kh_f, aL, T);

  // ---- Frechet mean iterations ----
  for (int it = 0; it < 2; ++it) {
    {
      GemmArgs p = ga0();                     // G = <mu_s, vh_s>
      p.A = mu_b + 1; p.sAb = (long)T * LD33; p.ldA = LD33;
      p.Bb = vh_b + 1; p.sBb = (long)T * LD33; p.ldB = LD33; p.transB = 1; p.bbf16 = 1;
      p.C = innerL; p.sCb = (long)T * T; p.ldC = T;
      p.M = T; p.N = T; p.K = NL;
      launch_gemm(stream, p, ZL);
    }
    hipLaunchKernelGGL(fweights_kernel, dim3(ZL * T), dim3(256), 0, stream,
                       innerL, aL, mu_f, vh_f, w1_b, w2_f, T);
    {
      GemmArgs p = ga0();                     // step_raw = w1 @ vh (all 33 dims)
      p.A = w1_b; p.sAb = (long)T * T; p.ldA = T;
      p.Bb = vh_b; p.sBb = (long)T * LD33; p.ldB = LD33; p.bbf16 = 1;
      p.C = step_f; p.sCb = (long)T * LD33; p.ldC = LD33;
      p.M = T; p.N = LD33; p.K = T;
      launch_gemm(stream, p, ZL);
    }
    hipLaunchKernelGGL(expmap_kernel, dim3((nZT + 127) / 128), dim3(128), 0, stream,
                       step_f, w2_f, mu_f, mu_b, nZT);
  }
  hipLaunchKernelGGL(logmap_kernel, dim3((nZT + 127) / 128), dim3(128), 0, stream,
                     mu_f, y_bf, T, HH, nZT);

  // ---- h_out = y_log @ lo_w + lo_b ----
  {
    GemmArgs p = ga0();
    p.A = y_bf; p.ldA = HH * LD33;
    p.Bf = lo_w; p.ldB = D; p.bias = lo_b;
    p.Cb = hout_bf; p.ldCb = D;
    p.M = M; p.N = D; p.K = HH * LD33;        // K = 132, guarded
    launch_gemm(stream, p, 1);
  }

  // ---- fusion projections e_f / h_f (f32 + bf16 halves of concat) ----
  {
    GemmArgs p = ga0();
    p.A = eout_bf; p.ldA = D; p.Bf = ue_w; p.ldB = D; p.bias = ue_b;
    p.C = ef_f; p.ldC = D; p.Cb = cat_bf; p.ldCb = 2 * D;
    p.M = M; p.N = D; p.K = D;
    launch_gemm(stream, p, 1);
  }
  {
    GemmArgs p = ga0();
    p.A = hout_bf; p.ldA = D; p.Bf = uh_w; p.ldB = D; p.bias = uh_b;
    p.C = hf_f; p.ldC = D; p.Cb = cat_bf + D; p.ldCb = 2 * D;
    p.M = M; p.N = D; p.K = D;
    launch_gemm(stream, p, 1);
  }
  {
    GemmArgs p = ga0();                       // g = tanh(cat @ gh_w + gh_b)
    p.A = cat_bf; p.ldA = 2 * D; p.Bf = gh_w; p.ldB = GH; p.bias = gh_b;
    p.C = gate_f; p.ldC = GH; p.act = 2;
    p.M = M; p.N = GH; p.K = 2 * D;
    launch_gemm(stream, p, 1);
  }
  hipLaunchKernelGGL(gate_kernel, dim3(M), dim3(64), 0, stream,
                     gate_f, go_w, go_b, ef_f, hf_f, x, x1_f, GH, D);

  // ---- FFN ----
  hipLaunchKernelGGL(ln_kernel, dim3(M), dim3(256), 0, stream, x1_f, ln2_g, ln2_b, h2_bf, D);
  {
    GemmArgs p = ga0();                       // gelu(h2 @ fi_w + fi_b)
    p.A = h2_bf; p.ldA = D; p.Bf = fi_w; p.ldB = FFN; p.bias = fi_b;
    p.Cb = ffn1_bf; p.ldCb = FFN; p.act = 1;
    p.M = M; p.N = FFN; p.K = D;
    launch_gemm(stream, p, 1);
  }
  {
    GemmArgs p = ga0();                       // out = x1 + ffn1 @ fo_w + fo_b
    p.A = ffn1_bf; p.ldA = FFN; p.Bf = fo_w; p.ldB = D; p.bias = fo_b;
    p.C = out; p.ldC = D; p.addC = x1_f;
    p.M = M; p.N = D; p.K = FFN;
    launch_gemm(stream, p, 1);
  }
}
